// LinearAttention_4947802325400
// MI455X (gfx1250) — matrix operand order repaired and checked by reading, not compiled, not run on hardware
//
#include <hip/hip_runtime.h>
#include <hip/hip_bf16.h>

// Problem constants
#define BB 4
#define NN 8192
#define DD 1024
#define HH 8
#define HD 128
#define M_TOT (BB * NN)         // 32768
#define N3D (3 * DD)            // 3072
#define NTILES (N3D / 128)      // 24
#define KCHUNKS (DD / 32)       // 32

typedef __attribute__((ext_vector_type(16))) __bf16 v16bf;
typedef __attribute__((ext_vector_type(8)))  float  v8f;
typedef __attribute__((ext_vector_type(4)))  unsigned int u32x4;
typedef __attribute__((ext_vector_type(8)))  unsigned int u32x8;

union BfFrag {
    v16bf v;
    uint4 q[2];
    unsigned int u[8];
    __bf16 h[16];
};

union Pk2 { __bf16 h[2]; unsigned int u; };
union Pk4 { __bf16 h[4]; uint2 u2; };

// ---------------------------------------------------------------------------
// CDNA5 Tensor Data Mover: one-instruction tile DMA global->LDS (TENSORcnt).
// D# groups per CDNA5 ISA. Groups 2/3 zero
// (2D tensors; tile_dim2..4 unused). Wave-uniform operands -> SGPRs.
// ---------------------------------------------------------------------------
__device__ __forceinline__ void tdm_load(unsigned lds_addr, unsigned long long ga,
                                         u32x8 g1) {
    u32x4 g0;
    g0[0] = 1u;                                   // count=1 (valid user D#)
    g0[1] = lds_addr;                             // LDS byte address
    g0[2] = (unsigned)ga;                         // global_addr[31:0]
    g0[3] = (unsigned)(ga >> 32) | 0x80000000u;   // global_addr[56:32], type=2
    u32x4 gz = {0u, 0u, 0u, 0u};
    asm volatile("tensor_load_to_lds %0, %1, %2, %3"
                 :: "s"(g0), "s"(g1), "s"(gz), "s"(gz)
                 : "memory");
}

// ---------------------------------------------------------------------------
// Kernel 1: weights fp32 -> bf16, pre-swizzled into the exact per-tile LDS
// image for the WMMA B operand: per (ktile 32 x ntile 128) a contiguous
// 2560-dword block laid out [n][k-pair], row stride 20 dwords.
// ---------------------------------------------------------------------------
__global__ void __launch_bounds__(256)
swz_w_kernel(const float* __restrict__ w, unsigned int* __restrict__ wswz) {
    const int tile = blockIdx.x;        // 0..KCHUNKS*NTILES-1
    const int tk = tile / NTILES;       // k-chunk (32 rows of w)
    const int tn = tile % NTILES;       // 128-col tile
    unsigned int* dst = wswz + (size_t)tile * 2560;
    for (int i = threadIdx.x; i < 2560; i += 256) {
        const int n = i / 20;
        const int p = i % 20;
        unsigned int val = 0;
        if (p < 16) {
            const int k = tk * 32 + 2 * p;
            const int c = tn * 128 + n;
            Pk2 pk;
            pk.h[0] = (__bf16)w[(size_t)k * N3D + c];
            pk.h[1] = (__bf16)w[(size_t)(k + 1) * N3D + c];
            val = pk.u;
        }
        dst[i] = val;
    }
}

// ---------------------------------------------------------------------------
// Kernel 2: LayerNorm (D=1024) + convert to bf16. One block (256 thr) per token.
// ---------------------------------------------------------------------------
__global__ void __launch_bounds__(256) ln_bf16_kernel(const float* __restrict__ x,
                                                      const float* __restrict__ gamma,
                                                      const float* __restrict__ beta,
                                                      __bf16* __restrict__ xn) {
    __shared__ float red[16];
    __shared__ float mv[2];
    const int row = blockIdx.x;
    const int t = threadIdx.x;
    const float* xr = x + (size_t)row * DD;
    float4 v = ((const float4*)xr)[t];
    float s  = v.x + v.y + v.z + v.w;
    float s2 = v.x * v.x + v.y * v.y + v.z * v.z + v.w * v.w;
#pragma unroll
    for (int off = 16; off > 0; off >>= 1) {
        s  += __shfl_down(s, off, 32);
        s2 += __shfl_down(s2, off, 32);
    }
    const int wv = t >> 5;
    if ((t & 31) == 0) { red[wv] = s; red[8 + wv] = s2; }
    __syncthreads();
    if (t == 0) {
        float ts = 0.f, ts2 = 0.f;
#pragma unroll
        for (int i = 0; i < 8; i++) { ts += red[i]; ts2 += red[8 + i]; }
        float mean = ts * (1.f / DD);
        float var  = ts2 * (1.f / DD) - mean * mean;
        mv[0] = mean;
        mv[1] = rsqrtf(var + 1e-5f);
    }
    __syncthreads();
    const float mean = mv[0], rstd = mv[1];
    float4 g = ((const float4*)gamma)[t];
    float4 b = ((const float4*)beta)[t];
    Pk4 pk;
    pk.h[0] = (__bf16)((v.x - mean) * rstd * g.x + b.x);
    pk.h[1] = (__bf16)((v.y - mean) * rstd * g.y + b.y);
    pk.h[2] = (__bf16)((v.z - mean) * rstd * g.z + b.z);
    pk.h[3] = (__bf16)((v.w - mean) * rstd * g.w + b.w);
    ((uint2*)(xn + (size_t)row * DD))[t] = pk.u2;
}

// ---------------------------------------------------------------------------
// Kernel 3: QKV GEMM [32768,1024] x [1024,3072], fused activation epilogue.
// 128x128 tile/block, 8 waves. BOTH operand tiles double-buffered in LDS via
// Tensor Data Mover (one tensor_load_to_lds per tile per chunk, wave 0 only).
// Main loop is pure LDS b128 fragment loads + WMMA.
// ---------------------------------------------------------------------------
__global__ void __launch_bounds__(256)
qkv_gemm_kernel(const __bf16* __restrict__ A,           // xn  [M_TOT, DD]
                const unsigned int* __restrict__ wswz,  // swizzled weights
                const unsigned char* __restrict__ mask, // [BB, NN]
                __bf16* __restrict__ qb,                // [BB,HH,NN,HD]
                __bf16* __restrict__ kb,
                __bf16* __restrict__ vb) {
    __shared__ __align__(16) __bf16 As[2][128 * 32];        // 2 x 8 KB
    __shared__ __align__(16) unsigned int BsU[2][128 * 20]; // 2 x 10 KB

    const int mBase = blockIdx.y * 128;
    const int tnIdx = blockIdx.x;
    const int nBase = tnIdx * 128;
    const int t    = threadIdx.x;
    const int lane = t & 31;
    const int wave = t >> 5;
    const int half = lane >> 4;
    const int l16  = lane & 15;

    // TDM descriptors (wave-uniform -> SGPRs)
    // A: 2D tile 32 (contig) x 128 rows of bf16, row stride 1024 elems
    const u32x8 g1A = {0x00010000u,      // data_size=1 (2B)
                       0x04000000u,      // tensor_dim0 = 1024 (low16<<16)
                       0x80000000u,      // tensor_dim1 = 32768 (low16<<16)
                       0x00200000u,      // tile_dim0 = 32 (<<16)
                       0x00000080u,      // tile_dim1 = 128
                       0x00000400u,      // tensor_dim0_stride = 1024
                       0u, 0u};
    // B: 1D contiguous 5120 bf16 (one pre-swizzled 10KB tile)
    const u32x8 g1B = {0x00010000u,      // data_size=1 (2B)
                       0x00000000u,      // tensor_dim0 low16 = 0
                       0x0001003Cu,      // tensor_dim0 hi16=0x3C, tensor_dim1=1
                       0x14000000u,      // tile_dim0 = 5120 (<<16)
                       0x00000001u,      // tile_dim1 = 1
                       0x00001400u,      // tensor_dim0_stride = 5120
                       0u, 0u};
    const unsigned long long gaA = (unsigned long long)(size_t)A +
                                   (size_t)mBase * (DD * 2);
    const unsigned long long gaB = (unsigned long long)(size_t)wswz;
    const unsigned ldsA[2] = {
        (unsigned)(size_t)(const void*)&As[0][0],
        (unsigned)(size_t)(const void*)&As[1][0]
    };
    const unsigned ldsB[2] = {
        (unsigned)(size_t)(const void*)&BsU[0][0],
        (unsigned)(size_t)(const void*)&BsU[1][0]
    };

    v8f acc[8];
#pragma unroll
    for (int i = 0; i < 8; i++) { v8f z = {}; acc[i] = z; }

    // prologue: DMA chunk 0 into buffer 0
    if (wave == 0) {
        tdm_load(ldsA[0], gaA, g1A);
        tdm_load(ldsB[0], gaB + (size_t)tnIdx * 10240, g1B);
    }

    int cur = 0;
    for (int kk = 0; kk < DD; kk += 32) {
        __builtin_amdgcn_s_wait_tensorcnt(0);
        __syncthreads();
        // DMA next chunk into the other buffer while computing this one
        if (kk + 32 < DD && wave == 0) {
            tdm_load(ldsA[cur ^ 1], gaA + (size_t)(kk + 32) * 2, g1A);
            tdm_load(ldsB[cur ^ 1],
                     gaB + ((size_t)(((kk + 32) >> 5) * NTILES + tnIdx)) * 10240,
                     g1B);
        }
        // compute: wave = one 16-row strip x 8 column tiles
        BfFrag a;
        {
            const uint4* s = (const uint4*)As[cur];
            const int base = (wave * 16 + l16) * 4 + half;
            a.q[0] = s[base];
            a.q[1] = s[base + 2];
        }
#pragma unroll
        for (int nt = 0; nt < 8; nt++) {
            BfFrag bf;
            const uint4* bs =
                (const uint4*)&BsU[cur][(nt * 16 + l16) * 20 + half * 4];
            bf.q[0] = bs[0];
            bf.q[1] = bs[2];
            acc[nt] = __builtin_amdgcn_wmma_f32_16x16x32_bf16(
                false, a.v, false, bf.v, (short)0, acc[nt], false, false);
        }
        __syncthreads();
        cur ^= 1;
    }

    // --- fused epilogue: 128-col block maps to one of q/k/v, one head ---
    const int sec = nBase >> 10;
    const int h   = (nBase & 1023) >> 7;
#pragma unroll
    for (int nt = 0; nt < 8; nt++) {
#pragma unroll
        for (int r = 0; r < 8; r++) {
            const int mrow = mBase + wave * 16 + r + half * 8;
            const int b = mrow >> 13;
            const int n = mrow & (NN - 1);
            const int hd = nt * 16 + l16;
            float val = acc[nt][r];
            const size_t dst = ((size_t)(b * HH + h) * NN + n) * HD + hd;
            if (sec == 0) {
                qb[dst] = (__bf16)(1.f / (1.f + __expf(-val)));
            } else if (sec == 1) {
                float kvv = tanhf(val);
                if (mask[b * NN + n]) kvv = 0.f;
                kb[dst] = (__bf16)kvv;
            } else {
                vb[dst] = (__bf16)val;
            }
        }
    }
}

// ---------------------------------------------------------------------------
// Kernel 4: zero the kv accumulator
// ---------------------------------------------------------------------------
__global__ void __launch_bounds__(256) zero_f32_kernel(float* p) {
    p[(size_t)blockIdx.x * 256 + threadIdx.x] = 0.f;
}

// ---------------------------------------------------------------------------
// Kernel 5: kv = k^T @ v per (b,h); split-K over 8 n-chunks, atomic reduce.
// ---------------------------------------------------------------------------
__global__ void __launch_bounds__(256)
kv_gemm_kernel(const __bf16* __restrict__ kb, const __bf16* __restrict__ vb,
               float* __restrict__ kvbuf) {
    __shared__ __align__(16) __bf16 KT[128 * 32];            // [d][n_local]
    __shared__ __align__(16) unsigned int VsU[128 * 20];     // [e][n-pair]

    const int bh = blockIdx.y;
    const int n0base = blockIdx.x * 1024;
    const int t    = threadIdx.x;
    const int lane = t & 31;
    const int wave = t >> 5;
    const int half = lane >> 4;
    const int l16  = lane & 15;

    const int knl = t >> 3;
    const int kdb = (t & 7) * 16;
    const int vp  = t >> 4;
    const int veb = (t & 15) * 8;

    v8f acc[8];
#pragma unroll
    for (int i = 0; i < 8; i++) { v8f z = {}; acc[i] = z; }

    BfFrag kr;
    uint4 vr0, vr1;
    {
        const uint4* g = (const uint4*)(kb + ((size_t)bh * NN + n0base + knl) * HD + kdb);
        kr.q[0] = g[0]; kr.q[1] = g[1];
        const uint4* g0 = (const uint4*)(vb + ((size_t)bh * NN + n0base + 2 * vp) * HD + veb);
        const uint4* g1 = (const uint4*)(vb + ((size_t)bh * NN + n0base + 2 * vp + 1) * HD + veb);
        vr0 = *g0; vr1 = *g1;
    }

    for (int n0 = n0base; n0 < n0base + 1024; n0 += 32) {
#pragma unroll
        for (int j = 0; j < 16; j++) KT[(kdb + j) * 32 + knl] = kr.h[j];
        {
            const __bf16* h0 = (const __bf16*)&vr0;
            const __bf16* h1 = (const __bf16*)&vr1;
#pragma unroll
            for (int j = 0; j < 8; j++) {
                Pk2 pk; pk.h[0] = h0[j]; pk.h[1] = h1[j];
                VsU[(veb + j) * 20 + vp] = pk.u;
            }
        }
        __syncthreads();
        if (n0 + 32 < n0base + 1024) {
            const uint4* g = (const uint4*)(kb + ((size_t)bh * NN + n0 + 32 + knl) * HD + kdb);
            kr.q[0] = g[0]; kr.q[1] = g[1];
            const uint4* g0 = (const uint4*)(vb + ((size_t)bh * NN + n0 + 32 + 2 * vp) * HD + veb);
            const uint4* g1 = (const uint4*)(vb + ((size_t)bh * NN + n0 + 32 + 2 * vp + 1) * HD + veb);
            vr0 = *g0; vr1 = *g1;
        }
        BfFrag a;
        {
            const uint4* s = (const uint4*)KT;
            const int base = (wave * 16 + l16) * 4 + half;
            a.q[0] = s[base];
            a.q[1] = s[base + 2];
        }
#pragma unroll
        for (int nt = 0; nt < 8; nt++) {
            BfFrag bf;
            const uint4* bs = (const uint4*)&VsU[(nt * 16 + l16) * 20 + half * 4];
            bf.q[0] = bs[0];
            bf.q[1] = bs[2];
            acc[nt] = __builtin_amdgcn_wmma_f32_16x16x32_bf16(
                false, a.v, false, bf.v, (short)0, acc[nt], false, false);
        }
        __syncthreads();
    }
#pragma unroll
    for (int nt = 0; nt < 8; nt++) {
#pragma unroll
        for (int r = 0; r < 8; r++) {
            const int drow = wave * 16 + r + half * 8;
            const int e = nt * 16 + l16;
            atomicAdd(&kvbuf[(size_t)bh * (HD * HD) + drow * HD + e], acc[nt][r]);
        }
    }
}

// ---------------------------------------------------------------------------
// Kernel 6: out = q @ kv per (b,h). kv staged once as bf16 in LDS, stride-68.
// ---------------------------------------------------------------------------
__global__ void __launch_bounds__(256)
out_gemm_kernel(const __bf16* __restrict__ qb, const float* __restrict__ kvbuf,
                float* __restrict__ out) {
    __shared__ __align__(16) unsigned int KVU[128 * 68];   // ~34 KB

    const int bh = blockIdx.y;
    const int b = bh >> 3;
    const int h = bh & 7;
    const int nbase = blockIdx.x * 128;
    const int t    = threadIdx.x;
    const int lane = t & 31;
    const int wave = t >> 5;
    const int half = lane >> 4;
    const int l16  = lane & 15;

    {
        const int e  = t >> 1;
        const int pb = (t & 1) * 32;
        const float* g = kvbuf + (size_t)bh * (HD * HD) + e;
#pragma unroll
        for (int j = 0; j < 32; j++) {
            const int p = pb + j;
            Pk2 pk;
            pk.h[0] = (__bf16)g[(2 * p) * HD];
            pk.h[1] = (__bf16)g[(2 * p + 1) * HD];
            KVU[e * 68 + p] = pk.u;
        }
    }
    __syncthreads();

    v8f acc[8];
#pragma unroll
    for (int i = 0; i < 8; i++) { v8f z = {}; acc[i] = z; }

    const int row = nbase + wave * 16 + l16;
    const __bf16* qrow = qb + ((size_t)bh * NN + row) * HD;
#pragma unroll
    for (int dc = 0; dc < 4; dc++) {
        BfFrag a;
        const uint4* g = (const uint4*)(qrow + dc * 32);
        a.q[0] = g[half];
        a.q[1] = g[2 + half];
#pragma unroll
        for (int nt = 0; nt < 8; nt++) {
            BfFrag bf;
            const uint4* bs =
                (const uint4*)&KVU[(nt * 16 + l16) * 68 + dc * 16 + half * 4];
            bf.q[0] = bs[0];
            bf.q[1] = bs[2];
            acc[nt] = __builtin_amdgcn_wmma_f32_16x16x32_bf16(
                false, a.v, false, bf.v, (short)0, acc[nt], false, false);
        }
    }
#pragma unroll
    for (int nt = 0; nt < 8; nt++) {
#pragma unroll
        for (int r = 0; r < 8; r++) {
            const int m = nbase + wave * 16 + r + half * 8;
            const int e = nt * 16 + l16;
            out[((size_t)b * NN + m) * DD + h * HD + e] = acc[nt][r];
        }
    }
}

// ---------------------------------------------------------------------------
// Host launch
// ---------------------------------------------------------------------------
extern "C" void kernel_launch(void* const* d_in, const int* in_sizes, int n_in,
                              void* d_out, int out_size, void* d_ws, size_t ws_size,
                              hipStream_t stream) {
    const float* x            = (const float*)d_in[0];
    const unsigned char* mask = (const unsigned char*)d_in[1];   // jnp.bool_
    const float* w            = (const float*)d_in[2];
    const float* gamma        = (const float*)d_in[3];
    const float* beta         = (const float*)d_in[4];
    float* out = (float*)d_out;

    char* ws = (char*)d_ws;
    const size_t SZ_XN   = (size_t)M_TOT * DD * 2;                  // 64 MB
    const size_t SZ_WSWZ = (size_t)KCHUNKS * NTILES * 2560 * 4;     // 7.5 MB
    const size_t SZ_QKV  = (size_t)M_TOT * DD * 2;                  // 64 MB each
    __bf16* xn        = (__bf16*)(ws);
    unsigned int* wsz = (unsigned int*)(ws + SZ_XN);
    __bf16* qb = (__bf16*)(ws + SZ_XN + SZ_WSWZ);
    __bf16* kb = (__bf16*)(ws + SZ_XN + SZ_WSWZ + SZ_QKV);
    __bf16* vb = (__bf16*)(ws + SZ_XN + SZ_WSWZ + 2 * SZ_QKV);
    float*  kv = (float*)(ws + SZ_XN + SZ_WSWZ + 3 * SZ_QKV);

    swz_w_kernel<<<dim3(KCHUNKS * NTILES), 256, 0, stream>>>(w, wsz);
    ln_bf16_kernel<<<dim3(M_TOT), 256, 0, stream>>>(x, gamma, beta, xn);
    qkv_gemm_kernel<<<dim3(NTILES, M_TOT / 128), 256, 0, stream>>>(
        xn, wsz, mask, qb, kb, vb);
    zero_f32_kernel<<<dim3((BB * HH * HD * HD) / 256), 256, 0, stream>>>(kv);
    kv_gemm_kernel<<<dim3(8, BB * HH), 256, 0, stream>>>(kb, vb, kv);
    out_gemm_kernel<<<dim3(NN / 128, BB * HH), 256, 0, stream>>>(qb, kv, out);
}
